// GuidedSAModuleMSG_78537771975382
// MI455X (gfx1250) — matrix operand order repaired and checked by reading, not compiled, not run on hardware
//
#include <hip/hip_runtime.h>
#include <hip/hip_bf16.h>

typedef __attribute__((ext_vector_type(16))) _Float16 v16h;
typedef __attribute__((ext_vector_type(8)))  _Float16 v8h;
typedef __attribute__((ext_vector_type(8)))  float    v8f;

#define B_SZ   2
#define N_PER  8192
#define M_PER  2048
#define C_IN   64
#define ROWS   256   // GEMM rows per workgroup = centersPerWG * nsample

static __device__ __forceinline__ unsigned packh2(float a, float b) {
  union { _Float16 h[2]; unsigned u; } q;
  q.h[0] = (_Float16)a; q.h[1] = (_Float16)b;
  return q.u;
}

// ---------------------------------------------------------------------------
// BN-fold + layout prep:  Wb[o*kp + k] = gamma*rsqrt(var+eps) * W[o][k'],
// K-major padded to kp; layer-1 permutes K so cols 0..63 = features,
// 64..66 = xyz (matches gather layout), rest zero.  bias = beta - mean*s.
// ---------------------------------------------------------------------------
__global__ void prep_weights_kernel(const float* __restrict__ W,
                                    const float* __restrict__ gamma,
                                    const float* __restrict__ beta,
                                    const float* __restrict__ mean,
                                    const float* __restrict__ var,
                                    int cin, int kp, int permute_xyz,
                                    _Float16* __restrict__ Wb,
                                    float* __restrict__ bias) {
  const int o = blockIdx.x;
  const float s = gamma[o] * rsqrtf(var[o] + 1e-3f);
  if (threadIdx.x == 0) bias[o] = beta[o] - mean[o] * s;
  for (int k = threadIdx.x; k < kp; k += blockDim.x) {
    float w = 0.0f;
    if (permute_xyz) {
      if (k < cin - 3)      w = W[o * cin + 3 + k];           // features
      else if (k < cin)     w = W[o * cin + (k - (cin - 3))]; // xyz
    } else if (k < cin) {
      w = W[o * cin + k];
    }
    Wb[o * kp + k] = (_Float16)(w * s);
  }
}

__global__ void copy_xyz_kernel(const float* __restrict__ src,
                                float* __restrict__ dst, int n) {
  int i = blockIdx.x * blockDim.x + threadIdx.x;
  if (i < n) dst[i] = src[i];
}

// ---------------------------------------------------------------------------
// One MLP layer on a 256-row LDS tile using v_wmma_f32_16x16x32_f16.
// nt-outer loop keeps B-fragments loop-invariant in registers for every
// layer shape; each wave then sweeps 2 M-tiles per B load.
// A layout (16-bit A 16x32): lane m=lane&15, g=lane>>4;
//   half h = K[16g + h] (same order as B).
// B layout (32x16): lane n=lane&15, g=lane>>4; half h = K[16g + h].
// C/D: vgpr r -> row = 16*mt + r + 8g, col = 16*nt + n.
// ---------------------------------------------------------------------------
template <int KP, int COUT, bool POOL, int NS>
__device__ __forceinline__ void mlp_layer(const _Float16* __restrict__ Wb,
                                          const float* __restrict__ bias,
                                          const _Float16* actIn,
                                          _Float16* actOut,
                                          unsigned* poolBits) {
  const int tid  = threadIdx.x;
  const int wave = tid >> 5;
  const int lane = tid & 31;
  const int g    = lane >> 4;
  const int n    = lane & 15;
  constexpr int nMT = ROWS / 16;   // 16 M-tiles
  constexpr int nNT = COUT / 16;
  constexpr int nKT = KP / 32;

  for (int nt = 0; nt < nNT; ++nt) {
    // B-fragments: loop-invariant across the M sweep
    v16h Bf[nKT];
#pragma unroll
    for (int kt = 0; kt < nKT; ++kt)
      Bf[kt] = *(const v16h*)(Wb + (nt * 16 + n) * KP + kt * 32 + 16 * g);

    const int col  = nt * 16 + n;
    const float bv = bias[col];

    for (int mt = wave; mt < nMT; mt += 8) {
      v8f acc = {};
#pragma unroll
      for (int kt = 0; kt < nKT; ++kt) {
        const _Float16* aRow = actIn + (mt * 16 + n) * KP + kt * 32;
        v8h aLo = *(const v8h*)(aRow + 16 * g);
        v8h aHi = *(const v8h*)(aRow + 16 * g + 8);
        v16h A;
#pragma unroll
        for (int i = 0; i < 8; ++i) { A[i] = aLo[i]; A[8 + i] = aHi[i]; }
        acc = __builtin_amdgcn_wmma_f32_16x16x32_f16(false, A, false, Bf[kt],
                                                     (short)0, acc, false, false);
      }
      if (POOL) {
        // all 8 rows of this lane (mt*16 + 8g + r) share one center:
        // reduce in registers, single LDS atomic per lane per tile.
        float m = acc[0] + bv;
#pragma unroll
        for (int r = 1; r < 8; ++r) m = fmaxf(m, acc[r] + bv);
        m = fmaxf(m, 0.0f);
        const int center = (mt * 16 + 8 * g) / NS;
        atomicMax(&poolBits[center * COUT + col], __float_as_uint(m));
      } else {
#pragma unroll
        for (int r = 0; r < 8; ++r) {
          const int row = mt * 16 + 8 * g + r;
          float x = fmaxf(acc[r] + bv, 0.0f);
          actOut[row * COUT + col] = (_Float16)x;
        }
      }
    }
  }
}

// ---------------------------------------------------------------------------
// Fused branch: ball-query -> gather -> 3x(WMMA GEMM + BN/ReLU) -> maxpool.
// ---------------------------------------------------------------------------
template <int NS, int C1, int C2, int C3>
__global__ __launch_bounds__(256) void sa_branch_kernel(
    const float* __restrict__ xyz, const float* __restrict__ features,
    const float* __restrict__ new_xyz,
    const _Float16* __restrict__ Wb1, const _Float16* __restrict__ Wb2,
    const _Float16* __restrict__ Wb3,
    const float* __restrict__ bias1, const float* __restrict__ bias2,
    const float* __restrict__ bias3,
    float* __restrict__ outF, float radius2, int colBase) {
  constexpr int CPW = ROWS / NS;   // centers per workgroup
  constexpr int K0P = 96;          // 67 -> padded to 3*32
  __shared__ alignas(32) _Float16 bufA[ROWS * K0P];
  __shared__ alignas(32) _Float16 bufB[ROWS * C1];
  __shared__ unsigned poolBits[CPW * C3];
  __shared__ int idxS[ROWS];
  __shared__ int cntS[CPW];

  const int tid    = threadIdx.x;
  const int wave   = tid >> 5;
  const int lane   = tid & 31;
  const int wgBase = blockIdx.x * CPW;

  for (int i = tid; i < CPW * C3; i += 256) poolBits[i] = 0u;

  // ---- ball query: one wave per center, 32-wide ordered append ----
  for (int cl = wave; cl < CPW; cl += 8) {
    const int cm   = wgBase + cl;
    const int bb   = cm / M_PER;
    const int base = bb * N_PER;
    const float cx = new_xyz[cm * 3 + 0];
    const float cy = new_xyz[cm * 3 + 1];
    const float cz = new_xyz[cm * 3 + 2];
    int cnt = 0;
    for (int it = 0; it < N_PER / 32 && cnt < NS; ++it) {
      const int li   = it * 32 + lane;
      const float dx = xyz[(size_t)(base + li) * 3 + 0] - cx;
      const float dy = xyz[(size_t)(base + li) * 3 + 1] - cy;
      const float dz = xyz[(size_t)(base + li) * 3 + 2] - cz;
      const bool inR = (dx * dx + dy * dy + dz * dz) < radius2;
#if __has_builtin(__builtin_amdgcn_ballot_w32)
      const unsigned m = __builtin_amdgcn_ballot_w32(inR);
#else
      const unsigned m = (unsigned)__ballot(inR);
#endif
      if (inR) {
        __builtin_prefetch(features + (size_t)(base + li) * C_IN, 0, 1);
        const int pos = cnt + __popc(m & ((1u << lane) - 1u));
        if (pos < NS) idxS[cl * NS + pos] = li;
      }
      cnt += __popc(m);
    }
    const int first = (cnt > 0) ? idxS[cl * NS] : 0;
    for (int s = lane; s < NS; s += 32)
      if (s >= cnt) idxS[cl * NS + s] = first;
    if (lane == 0) cntS[cl] = cnt;
  }
  __syncthreads();

  // ---- gather into LDS tile: cols 0..63 feat, 64..66 xyz-center, pad 0 ----
  {
    const int cl = tid / NS;
    const int cm = wgBase + cl;
    const int bb = cm / M_PER;
    unsigned* dst = (unsigned*)&bufA[tid * K0P];
    if (cntS[cl] == 0) {
#pragma unroll
      for (int i = 0; i < K0P / 2; ++i) dst[i] = 0u;
    } else {
      const int p = bb * N_PER + idxS[tid];
      const float4* f4 = (const float4*)(features + (size_t)p * C_IN);
#pragma unroll
      for (int i = 0; i < C_IN / 4; ++i) {
        const float4 v = f4[i];
        dst[2 * i + 0] = packh2(v.x, v.y);
        dst[2 * i + 1] = packh2(v.z, v.w);
      }
      const float cx = new_xyz[cm * 3 + 0];
      const float cy = new_xyz[cm * 3 + 1];
      const float cz = new_xyz[cm * 3 + 2];
      dst[32] = packh2(xyz[(size_t)p * 3 + 0] - cx, xyz[(size_t)p * 3 + 1] - cy);
      dst[33] = packh2(xyz[(size_t)p * 3 + 2] - cz, 0.0f);
#pragma unroll
      for (int i = 34; i < K0P / 2; ++i) dst[i] = 0u;
    }
  }
  __syncthreads();

  mlp_layer<K0P, C1, false, NS>(Wb1, bias1, bufA, bufB, nullptr);
  __syncthreads();
  mlp_layer<C1, C2, false, NS>(Wb2, bias2, bufB, bufA, nullptr);
  __syncthreads();
  mlp_layer<C2, C3, true, NS>(Wb3, bias3, bufA, nullptr, poolBits);
  __syncthreads();

  for (int i = tid; i < CPW * C3; i += 256) {
    const int cl = i / C3, col = i % C3;
    outF[(size_t)(wgBase + cl) * 256 + colBase + col] =
        __uint_as_float(poolBits[i]);
  }
}

// ---------------------------------------------------------------------------
extern "C" void kernel_launch(void* const* d_in, const int* in_sizes, int n_in,
                              void* d_out, int out_size, void* d_ws,
                              size_t ws_size, hipStream_t stream) {
  (void)in_sizes; (void)n_in; (void)out_size; (void)ws_size;
  const float* xyz      = (const float*)d_in[0];
  const float* new_xyz  = (const float*)d_in[2];
  const float* features = (const float*)d_in[4];
  auto P = [&](int i) { return (const float*)d_in[i]; };

  char* ws = (char*)d_ws;
  _Float16* wb11 = (_Float16*)(ws + 0);       //  64x96
  _Float16* wb12 = (_Float16*)(ws + 12288);   //  64x64
  _Float16* wb13 = (_Float16*)(ws + 20480);   // 128x64
  _Float16* wb21 = (_Float16*)(ws + 36864);   //  64x96
  _Float16* wb22 = (_Float16*)(ws + 49152);   //  96x64
  _Float16* wb23 = (_Float16*)(ws + 61440);   // 128x96
  float* b11 = (float*)(ws + 86016);
  float* b12 = (float*)(ws + 86272);
  float* b13 = (float*)(ws + 86528);
  float* b21 = (float*)(ws + 87040);
  float* b22 = (float*)(ws + 87296);
  float* b23 = (float*)(ws + 87680);

  // fold BN into f16 weights (K-major, padded, layer-1 K permuted)
  prep_weights_kernel<<<64, 64, 0, stream>>>(P(5),  P(6),  P(7),  P(8),  P(9),  67, 96, 1, wb11, b11);
  prep_weights_kernel<<<64, 64, 0, stream>>>(P(10), P(11), P(12), P(13), P(14), 64, 64, 0, wb12, b12);
  prep_weights_kernel<<<128,64, 0, stream>>>(P(15), P(16), P(17), P(18), P(19), 64, 64, 0, wb13, b13);
  prep_weights_kernel<<<64, 64, 0, stream>>>(P(20), P(21), P(22), P(23), P(24), 67, 96, 1, wb21, b21);
  prep_weights_kernel<<<96, 64, 0, stream>>>(P(25), P(26), P(27), P(28), P(29), 64, 64, 0, wb22, b22);
  prep_weights_kernel<<<128,64, 0, stream>>>(P(30), P(31), P(32), P(33), P(34), 96, 96, 0, wb23, b23);

  // output: [new_xyz (12288 f32)] [new_features (4096 x 256 f32)]
  copy_xyz_kernel<<<48, 256, 0, stream>>>(new_xyz, (float*)d_out, B_SZ * M_PER * 3);
  float* outF = (float*)d_out + B_SZ * M_PER * 3;

  // branch 1: r=0.4, ns=16, MLP 64-64-128 -> cols [0,128)
  sa_branch_kernel<16, 64, 64, 128><<<(B_SZ * M_PER) / 16, 256, 0, stream>>>(
      xyz, features, new_xyz, wb11, wb12, wb13, b11, b12, b13, outF, 0.16f, 0);
  // branch 2: r=0.8, ns=32, MLP 64-96-128 -> cols [128,256)
  sa_branch_kernel<32, 64, 96, 128><<<(B_SZ * M_PER) / 8, 256, 0, stream>>>(
      xyz, features, new_xyz, wb21, wb22, wb23, b21, b22, b23, outF, 0.64f, 128);
}
